// EGCL_72361609003289
// MI455X (gfx1250) — matrix operand order repaired and checked by reading, not compiled, not run on hardware
//
#include <hip/hip_runtime.h>
#include <hip/hip_bf16.h>

// EGNN layer fused kernel for gfx1250 (MI455X).
// One workgroup per receiver node; all edge MLPs via v_wmma_f32_16x16x32_bf16.

#define NNODES 384
#define VDIM   32
#define HDIM   128
#define UDIM   256
#define CHUNK  32
#define NCHUNK (NNODES / CHUNK)

typedef __attribute__((ext_vector_type(16))) __bf16       v16bf;
typedef __attribute__((ext_vector_type(8)))  float        v8f;
typedef __attribute__((ext_vector_type(4)))  unsigned int u32x4;

// Global-address-space views so laundered pointers still lower to
// global_load_b128 (FLAT loads would also consume DScnt / the LDS path).
typedef __attribute__((address_space(1))) const unsigned short g_ushort;
typedef __attribute__((address_space(1))) const u32x4          g_u32x4;

union FragBF { u32x4 q[2]; v16bf v; };

__device__ __forceinline__ unsigned short f2bf(float f) {
  unsigned int u = __float_as_uint(f);
  u += 0x7fffu + ((u >> 16) & 1u);       // round-to-nearest-even
  return (unsigned short)(u >> 16);
}
__device__ __forceinline__ float bf2f(unsigned short h) {
  return __uint_as_float(((unsigned int)h) << 16);
}
__device__ __forceinline__ float silu_f(float x) { return x / (1.f + __expf(-x)); }
__device__ __forceinline__ v8f vzero() {
  v8f z;
  #pragma unroll
  for (int i = 0; i < 8; ++i) z[i] = 0.f;
  return z;
}

// Make a pointer opaque to LICM so weight loads stay inside the chunk loop
// (prevents hoist-then-spill of all B fragments to scratch).
#define LAUNDER(p) asm volatile("" : "+s"(p))

// ---------------------------------------------------------------------------
// Weight repack: f32 [K x Nout] row-major -> bf16 WMMA B-fragment blocks.
// out[((nt*KT + kt)*32 + lane)*16 + i] = W[kt*32 + (i<8 ? 8*hi + i : 16 + 8*hi + i-8)][nt*16 + (lane&15)], hi = lane>=16
// (B layout matches A: lanes 0-15 K=0..7,16..23; lanes 16-31 K=8..15,24..31)
// ---------------------------------------------------------------------------
__global__ void repack_w(const float* __restrict__ W, unsigned short* __restrict__ out,
                         int K, int Nout) {
  int KT = K >> 5, NT = Nout >> 4;
  int total = KT * NT * 512;
  for (int idx = blockIdx.x * blockDim.x + threadIdx.x; idx < total;
       idx += gridDim.x * blockDim.x) {
    int i    = idx & 15;
    int lane = (idx >> 4) & 31;
    int t    = idx >> 9;               // nt*KT + kt
    int kt   = t % KT, nt = t / KT;
    int k = kt * 32 + ((i < 8) ? i : i + 8) + ((lane >= 16) ? 8 : 0);
    int n = nt * 16 + (lane & 15);
    out[idx] = f2bf(W[(size_t)k * Nout + n]);
  }
}

// ---------------------------------------------------------------------------
// [32 x K] (LDS bf16, row-major, stride lda) x [K x 256] (packed B frags) ->
// silu(scale * .) -> [32 x 256] LDS bf16 (stride 256).
// Wave w owns M-tiles {0,1} x N-tiles {2w, 2w+1}.
// ---------------------------------------------------------------------------
template <int KT>
__device__ __forceinline__ void gemm256(const unsigned short* aL, int lda,
                                        g_ushort* wp,
                                        unsigned short* dL, float scale,
                                        int wave, int lane) {
  const int lh = lane & 15;
  const int hi = lane >> 4;            // 0 for lanes 0-15, 1 for 16-31
  v8f acc[2][2];
  #pragma unroll
  for (int mt = 0; mt < 2; ++mt)
    #pragma unroll
    for (int j = 0; j < 2; ++j) acc[mt][j] = vzero();

  #pragma unroll
  for (int kt = 0; kt < KT; ++kt) {
    FragBF a[2], b[2];
    #pragma unroll
    for (int mt = 0; mt < 2; ++mt) {
      // A layout: halfs 0-7 -> K=kb..kb+7, halfs 8-15 -> K=kb+16..kb+23, kb=kt*32+8*hi
      const u32x4* p = (const u32x4*)(aL + (mt * 16 + lh) * lda + kt * 32 + hi * 8);
      a[mt].q[0] = p[0];
      a[mt].q[1] = p[2];               // +16 halfs = +2 u32x4
    }
    #pragma unroll
    for (int j = 0; j < 2; ++j) {
      int nt = wave * 2 + j;
      g_u32x4* q = (g_u32x4*)(wp + (((size_t)nt * KT + kt) * 32 + lane) * 16);
      b[j].q[0] = q[0];
      b[j].q[1] = q[1];
    }
    #pragma unroll
    for (int mt = 0; mt < 2; ++mt)
      #pragma unroll
      for (int j = 0; j < 2; ++j)
        acc[mt][j] = __builtin_amdgcn_wmma_f32_16x16x32_bf16(
            false, a[mt].v, false, b[j].v, (short)0, acc[mt][j], false, false);
  }
  // C/D layout: elem e -> M = e + 8*hi, N = lane&15
  #pragma unroll
  for (int mt = 0; mt < 2; ++mt)
    #pragma unroll
    for (int j = 0; j < 2; ++j) {
      int n = (wave * 2 + j) * 16 + lh;
      #pragma unroll
      for (int e = 0; e < 8; ++e) {
        int m = mt * 16 + e + hi * 8;
        dL[m * UDIM + n] = f2bf(silu_f(acc[mt][j][e] * scale));
      }
    }
}

// [32 x 256] x [256 x 32] + bias -> f32 [32 x 32] (phi_x). Waves 0-3 only.
__device__ __forceinline__ void gemmV(const unsigned short* aL,
                                      g_ushort* wp,
                                      const float* __restrict__ bias,
                                      float* dL, int wave, int lane) {
  if (wave < 4) {
    const int lh = lane & 15, hi = lane >> 4;
    int mt = wave >> 1, nt = wave & 1;
    v8f acc = vzero();
    #pragma unroll
    for (int kt = 0; kt < 8; ++kt) {
      FragBF a, b;
      const u32x4* p = (const u32x4*)(aL + (mt * 16 + lh) * UDIM + kt * 32 + hi * 8);
      a.q[0] = p[0];
      a.q[1] = p[2];
      g_u32x4* q = (g_u32x4*)(wp + (((size_t)nt * 8 + kt) * 32 + lane) * 16);
      b.q[0] = q[0];
      b.q[1] = q[1];
      acc = __builtin_amdgcn_wmma_f32_16x16x32_bf16(false, a.v, false, b.v,
                                                    (short)0, acc, false, false);
    }
    int n = nt * 16 + lh;
    float bn = bias[n];
    #pragma unroll
    for (int e = 0; e < 8; ++e) {
      int m = mt * 16 + e + hi * 8;
      dL[m * VDIM + n] = acc[e] + bn;
    }
  }
}

// ---------------------------------------------------------------------------
// Main fused kernel: one block (256 threads / 8 waves) per receiver node r.
// ---------------------------------------------------------------------------
__global__ void __launch_bounds__(256) egcl_main(
    const float* __restrict__ nv, const float* __restrict__ nf,
    const unsigned short* We1p, const unsigned short* We2p,
    const unsigned short* Wx1p, const unsigned short* Wx2p,
    const unsigned short* Wxlp,
    const float* __restrict__ bxl, const float* __restrict__ Winf,
    const float* __restrict__ Wh1, const float* __restrict__ Wh2,
    const float* __restrict__ Whl,
    float* __restrict__ outv, float* __restrict__ outf) {
  __shared__ unsigned short sF[CHUNK * 288];   // edge_feat; reused (stride 256) for x2
  __shared__ unsigned short sA[CHUNK * UDIM];  // act1 / x1
  __shared__ unsigned short sM[CHUNK * UDIM];  // m_ij
  __shared__ float sVec[CHUNK * VDIM * 3];
  __shared__ float sInv[CHUNK * VDIM];
  __shared__ float sPhiX[CHUNK * VDIM];
  __shared__ float sE[CHUNK];
  __shared__ float sRV[VDIM * 3];
  __shared__ float sRF[HDIM];
  __shared__ float sX[UDIM + HDIM];
  __shared__ float sH1[UDIM];
  __shared__ float sH2[UDIM];

  const int r = blockIdx.x;
  const int tid = threadIdx.x;
  const int wave = tid >> 5, lane = tid & 31;

  if (tid < VDIM * 3) sRV[tid] = nv[r * (VDIM * 3) + tid];
  if (tid < HDIM)     sRF[tid] = nf[r * HDIM + tid];
  __syncthreads();

  float macc = 0.f;  // m_i[r][tid]
  float sacc = 0.f;  // shift accumulator for (v,c) = tid (tid < 96)

  g_ushort* we1 = (g_ushort*)We1p;
  g_ushort* we2 = (g_ushort*)We2p;
  g_ushort* wx1 = (g_ushort*)Wx1p;
  g_ushort* wx2 = (g_ushort*)Wx2p;
  g_ushort* wxl = (g_ushort*)Wxlp;

  for (int ch = 0; ch < NCHUNK; ++ch) {
    // --- prologue: vectors = x[r]-x[s], len2, 1/(1+len), edge features ------
    for (int idx = tid; idx < CHUNK * VDIM; idx += 256) {
      int m = idx >> 5, v = idx & 31;
      int s = ch * CHUNK + m;
      float dx = sRV[v * 3 + 0] - nv[((size_t)s * VDIM + v) * 3 + 0];
      float dy = sRV[v * 3 + 1] - nv[((size_t)s * VDIM + v) * 3 + 1];
      float dz = sRV[v * 3 + 2] - nv[((size_t)s * VDIM + v) * 3 + 2];
      float l2 = dx * dx + dy * dy + dz * dz;
      sVec[idx * 3 + 0] = dx; sVec[idx * 3 + 1] = dy; sVec[idx * 3 + 2] = dz;
      sInv[idx] = 1.f / (1.f + sqrtf(fmaxf(l2, 1e-20f)));
      sF[m * 288 + v] = f2bf(l2);
    }
    for (int idx = tid; idx < CHUNK * HDIM; idx += 256) {
      int m = idx >> 7, h = idx & 127;
      int s = ch * CHUNK + m;
      sF[m * 288 + 32 + h]  = f2bf(nf[(size_t)s * HDIM + h]);
      sF[m * 288 + 160 + h] = f2bf(sRF[h]);
    }
    __syncthreads();

    // --- phi_e ---------------------------------------------------------------
    LAUNDER(we1);
    gemm256<9>(sF, 288, we1, sA, 0.05892557f /*1/sqrt(288)*/, wave, lane);
    __syncthreads();
    LAUNDER(we2);
    gemm256<8>(sA, 256, we2, sM, 0.0625f, wave, lane);
    __syncthreads();

    // --- gate e = sigmoid(m_ij @ Winf / 16), mask self edge ------------------
    if (tid < CHUNK) {
      int m = tid, s = ch * CHUNK + m;
      float dot = 0.f;
      for (int k = 0; k < UDIM; ++k) dot += bf2f(sM[m * UDIM + k]) * Winf[k];
      sE[m] = (s == r) ? 0.f : 1.f / (1.f + __expf(-dot * 0.0625f));
    }
    __syncthreads();

    // --- m_i accumulation (registers, thread tid owns channel tid) ----------
    {
      float a = 0.f;
      for (int m = 0; m < CHUNK; ++m) a += bf2f(sM[m * UDIM + tid]) * sE[m];
      macc += a;
    }

    // --- phi_x MLP -----------------------------------------------------------
    LAUNDER(wx1);
    gemm256<8>(sM, 256, wx1, sA, 0.0625f, wave, lane);
    __syncthreads();
    LAUNDER(wx2);
    gemm256<8>(sA, 256, wx2, sF /*reused, stride 256*/, 0.0625f, wave, lane);
    __syncthreads();
    LAUNDER(wxl);
    gemmV(sF, wxl, bxl, sPhiX, wave, lane);
    __syncthreads();

    // --- shift accumulation (self edge has vec==0, auto-masked) --------------
    if (tid < VDIM * 3) {
      int v = tid / 3, c = tid % 3;
      float a = 0.f;
      for (int m = 0; m < CHUNK; ++m)
        a += sPhiX[m * VDIM + v] * sVec[(m * VDIM + v) * 3 + c] * sInv[m * VDIM + v];
      sacc += a;
    }
    __syncthreads();  // protect LDS buffers before next chunk prologue
  }

  // --- vectors_out = x[r] + shifts/(N-1) -------------------------------------
  if (tid < VDIM * 3) outv[(size_t)r * (VDIM * 3) + tid] = sRV[tid] + sacc * (1.f / 383.f);

  // --- phi_h on [m_i | h_r] (384 -> 256 -> 256 -> 128), VALU GEMV chain ------
  sX[tid] = macc;
  if (tid < HDIM) sX[UDIM + tid] = sRF[tid];
  __syncthreads();
  {
    float a1 = 0.f;
    for (int k = 0; k < UDIM + HDIM; ++k) a1 += sX[k] * Wh1[(size_t)k * UDIM + tid];
    sH1[tid] = silu_f(a1 * 0.051031036f /*1/sqrt(384)*/);
  }
  __syncthreads();
  {
    float a2 = 0.f;
    for (int k = 0; k < UDIM; ++k) a2 += sH1[k] * Wh2[(size_t)k * UDIM + tid];
    sH2[tid] = silu_f(a2 * 0.0625f);
  }
  __syncthreads();
  if (tid < HDIM) {
    float a3 = 0.f;
    for (int k = 0; k < UDIM; ++k) a3 += sH2[k] * Whl[(size_t)k * HDIM + tid];
    outf[(size_t)r * HDIM + tid] = a3 * 0.0625f + sRF[tid];
  }
}

// ---------------------------------------------------------------------------
extern "C" void kernel_launch(void* const* d_in, const int* in_sizes, int n_in,
                              void* d_out, int out_size, void* d_ws, size_t ws_size,
                              hipStream_t stream) {
  const float* nv   = (const float*)d_in[0];
  const float* nf   = (const float*)d_in[1];
  const float* We1  = (const float*)d_in[2];
  const float* We2  = (const float*)d_in[3];
  const float* Wx1  = (const float*)d_in[4];
  const float* Wx2  = (const float*)d_in[5];
  const float* Wxl  = (const float*)d_in[6];
  const float* bxl  = (const float*)d_in[7];
  const float* Winf = (const float*)d_in[8];
  const float* Wh1  = (const float*)d_in[9];
  const float* Wh2  = (const float*)d_in[10];
  const float* Whl  = (const float*)d_in[11];

  unsigned short* ws   = (unsigned short*)d_ws;
  unsigned short* We1p = ws;               // 9*16*512  = 73728 halfs
  unsigned short* We2p = We1p + 73728;     // 8*16*512  = 65536
  unsigned short* Wx1p = We2p + 65536;
  unsigned short* Wx2p = Wx1p + 65536;
  unsigned short* Wxlp = Wx2p + 65536;     // 8*2*512   = 8192

  auto rp = [&](const float* W, unsigned short* o, int K, int Nout) {
    int total = (K >> 5) * (Nout >> 4) * 512;
    repack_w<<<(total + 255) / 256, 256, 0, stream>>>(W, o, K, Nout);
  };
  rp(We1, We1p, 288, 256);
  rp(We2, We2p, 256, 256);
  rp(Wx1, Wx1p, 256, 256);
  rp(Wx2, Wx2p, 256, 256);
  rp(Wxl, Wxlp, 256, 32);

  float* outv = (float*)d_out;
  float* outf = outv + (size_t)NNODES * VDIM * 3;
  egcl_main<<<NNODES, 256, 0, stream>>>(nv, nf, We1p, We2p, Wx1p, Wx2p, Wxlp,
                                        bxl, Winf, Wh1, Wh2, Whl, outv, outf);
}
